// ClusterGCN_88923002897065
// MI455X (gfx1250) — matrix operand order repaired and checked by reading, not compiled, not run on hardware
//
#include <hip/hip_runtime.h>

#define NNODES 50000
#define FEAT   128
#define EPSBN  1e-5f

typedef __bf16 bf16_t;
typedef bf16_t v16bf __attribute__((ext_vector_type(16)));
typedef bf16_t v8bf  __attribute__((ext_vector_type(8)));
typedef float  v8f   __attribute__((ext_vector_type(8)));

static __device__ __forceinline__ unsigned short f2bf(float f) {
  unsigned int u = __float_as_uint(f);
  u += 0x7FFFu + ((u >> 16) & 1u);           // round-to-nearest-even
  return (unsigned short)(u >> 16);
}

// ---- transpose + convert a 128x128 fp32 weight to N-major bf16 ----
__global__ __launch_bounds__(256) void k_wt_bf16(const float* __restrict__ W,
                                                 unsigned short* __restrict__ Wt) {
  int idx = blockIdx.x * 256 + threadIdx.x;          // idx = n*128 + k
  if (idx < FEAT * FEAT) {
    int n = idx >> 7, k = idx & 127;
    Wt[idx] = f2bf(W[k * FEAT + n]);
  }
}

// ---- fp32 -> bf16 elementwise (4 elems/thread) ----
__global__ __launch_bounds__(256) void k_f32_to_bf16(const float* __restrict__ src,
                                                     unsigned short* __restrict__ dst,
                                                     int n4) {
  int i = blockIdx.x * 256 + threadIdx.x;
  if (i < n4) {
    float4 v = ((const float4*)src)[i];
    uint2 o;
    o.x = (unsigned)f2bf(v.x) | ((unsigned)f2bf(v.y) << 16);
    o.y = (unsigned)f2bf(v.z) | ((unsigned)f2bf(v.w) << 16);
    ((uint2*)dst)[i] = o;
  }
}

// ---- WMMA GEMM: C[M x 128] = A_bf16[M x 128] @ W (Wt is N-major bf16) ----
__global__ __launch_bounds__(256) void k_gemm_wmma(const unsigned short* __restrict__ A,
                                                   const unsigned short* __restrict__ Wt,
                                                   float* __restrict__ C, int M) {
  __shared__ unsigned short ldsW[FEAT * FEAT];       // 32 KB, LDS offset 0

  // Defeat "LDS global has no stores -> loads are undef" folding: a store the
  // compiler cannot prove dead (M is runtime; never equals INT_MAX here).
  if (M == 0x7FFFFFFF) ldsW[threadIdx.x] = (unsigned short)M;

  {
    // CDNA5 async global->LDS staging: 2048 x 16B chunks, 8 per thread.
    // LDS address comes from the VDST operand; ldsW is the only __shared__ -> offset 0.
#pragma unroll
    for (int j = 0; j < 8; ++j) {
      unsigned off = (unsigned)threadIdx.x * 16u + (unsigned)j * 4096u;
      unsigned long long ga = (unsigned long long)Wt + off;
      asm volatile("global_load_async_to_lds_b128 %0, %1, off"
                   :: "v"(off), "v"(ga) : "memory");
    }
    asm volatile("s_wait_asynccnt 0x0" ::: "memory");
  }
  __syncthreads();

  const int wave = threadIdx.x >> 5;
  const int lane = threadIdx.x & 31;
  const int half = lane >> 4;
  const int m16  = lane & 15;
  const int rowTile = blockIdx.x * 8 + wave;
  if (rowTile * 16 >= M) return;                     // wave-uniform: EXEC stays all-ones

  const unsigned short* Arow = A + (size_t)(rowTile * 16 + m16) * FEAT;

  v8f acc[8];
#pragma unroll
  for (int i = 0; i < 8; ++i)
#pragma unroll
    for (int j = 0; j < 8; ++j) acc[i][j] = 0.0f;

#pragma unroll
  for (int kt = 0; kt < 4; ++kt) {
    const int kb = kt * 32;
    // A fragment: lane half selects contiguous K run of 16 (matches B)
    v8bf a0 = *(const v8bf*)(Arow + kb + half * 16);
    v8bf a1 = *(const v8bf*)(Arow + kb + half * 16 + 8);
    v16bf a;
#pragma unroll
    for (int i = 0; i < 8; ++i) { a[i] = a0[i]; a[i + 8] = a1[i]; }

    // Batch all 8 B fragments first so the 8 WMMAs issue back-to-back
    v16bf bfr[8];
#pragma unroll
    for (int nt = 0; nt < 8; ++nt)
      bfr[nt] = *(const v16bf*)(&ldsW[(size_t)(nt * 16 + m16) * FEAT + kb + half * 16]);
#pragma unroll
    for (int nt = 0; nt < 8; ++nt)
      acc[nt] = __builtin_amdgcn_wmma_f32_16x16x32_bf16(
          false, a, false, bfr[nt], (short)0, acc[nt], false, false);
  }

  // C/D layout: VGPR j -> M = half*8 + j, N = nt*16 + m16
  float* Crow = C + (size_t)(rowTile * 16 + half * 8) * FEAT + m16;
#pragma unroll
  for (int nt = 0; nt < 8; ++nt)
#pragma unroll
    for (int j = 0; j < 8; ++j)
      Crow[(size_t)j * FEAT + nt * 16] = acc[nt][j];
}

// ---- init agg rows to per-feature bias ----
__global__ __launch_bounds__(256) void k_init_bias(float* __restrict__ agg,
                                                   const float* __restrict__ bias, int n4) {
  int i = blockIdx.x * 256 + threadIdx.x;
  if (i < n4) {
    int f0 = (i * 4) & 127;
    float4 o = make_float4(bias[f0], bias[f0 + 1], bias[f0 + 2], bias[f0 + 3]);
    ((float4*)agg)[i] = o;
  }
}

// ---- edge scatter-add: one wave per (batch, edge), 128 feats via float4/lane ----
__global__ __launch_bounds__(256) void k_scatter_vec(const float* __restrict__ sup,
                                                     const int* __restrict__ ei,
                                                     float* __restrict__ agg,
                                                     int E, int total) {
  int gid = blockIdx.x * 256 + threadIdx.x;
  int ew = gid >> 5;
  if (ew >= total) return;
  int lane = gid & 31;
  int b = ew / E;
  int e = ew - b * E;
  int row = ei[e];
  int col = ei[E + e];
  const float4 s = ((const float4*)(sup + ((size_t)b * NNODES + row) * FEAT))[lane];
  float* d = agg + ((size_t)b * NNODES + col) * FEAT + lane * 4;
  atomicAdd(d + 0, s.x);
  atomicAdd(d + 1, s.y);
  atomicAdd(d + 2, s.z);
  atomicAdd(d + 3, s.w);
}

__global__ __launch_bounds__(256) void k_zero_stats(float* __restrict__ gsum,
                                                    float* __restrict__ gsumsq) {
  int t = threadIdx.x;
  if (t < FEAT) { gsum[t] = 0.0f; gsumsq[t] = 0.0f; }
}

// ---- per-feature sum / sumsq (stride keeps f = tid % 128 fixed) ----
__global__ __launch_bounds__(256) void k_bn_stats(const float* __restrict__ h,
                                                  float* __restrict__ gsum,
                                                  float* __restrict__ gsumsq, int n) {
  __shared__ float ls[256], ls2[256];
  int tid = threadIdx.x;
  int stride = gridDim.x * 256;
  float s = 0.0f, s2 = 0.0f;
  for (int idx = blockIdx.x * 256 + tid; idx < n; idx += stride) {
    float v = h[idx];
    s += v; s2 += v * v;
  }
  ls[tid] = s; ls2[tid] = s2;
  __syncthreads();
  if (tid < FEAT) {
    atomicAdd(&gsum[tid], ls[tid] + ls[tid + FEAT]);
    atomicAdd(&gsumsq[tid], ls2[tid] + ls2[tid + FEAT]);
  }
}

__global__ __launch_bounds__(128) void k_bn_finalize(const float* __restrict__ gsum,
                                                     const float* __restrict__ gsumsq,
                                                     const float* __restrict__ gamma,
                                                     const float* __restrict__ beta,
                                                     float* __restrict__ scale,
                                                     float* __restrict__ shift, float invM) {
  int f = threadIdx.x;
  if (f < FEAT) {
    float m = gsum[f] * invM;
    float var = gsumsq[f] * invM - m * m;
    float sc = gamma[f] * rsqrtf(var + EPSBN);
    scale[f] = sc;
    shift[f] = beta[f] - m * sc;
  }
}

// ---- BN + ReLU, emit bf16 operand for next WMMA GEMM ----
__global__ __launch_bounds__(256) void k_bn_relu_bf16(const float* __restrict__ agg,
                                                      const float* __restrict__ scale,
                                                      const float* __restrict__ shift,
                                                      unsigned short* __restrict__ out, int n4) {
  int i = blockIdx.x * 256 + threadIdx.x;
  if (i < n4) {
    int f0 = (i * 4) & 127;
    float4 v = ((const float4*)agg)[i];
    float r0 = fmaxf(v.x * scale[f0 + 0] + shift[f0 + 0], 0.0f);
    float r1 = fmaxf(v.y * scale[f0 + 1] + shift[f0 + 1], 0.0f);
    float r2 = fmaxf(v.z * scale[f0 + 2] + shift[f0 + 2], 0.0f);
    float r3 = fmaxf(v.w * scale[f0 + 3] + shift[f0 + 3], 0.0f);
    uint2 o;
    o.x = (unsigned)f2bf(r0) | ((unsigned)f2bf(r1) << 16);
    o.y = (unsigned)f2bf(r2) | ((unsigned)f2bf(r3) << 16);
    ((uint2*)out)[i] = o;
  }
}

// ---- BN + ReLU in-place fp32 (feeds the fp32 M-V product of layer 3) ----
__global__ __launch_bounds__(256) void k_bn_relu_f32(float* __restrict__ agg,
                                                     const float* __restrict__ scale,
                                                     const float* __restrict__ shift, int n4) {
  int i = blockIdx.x * 256 + threadIdx.x;
  if (i < n4) {
    int f0 = (i * 4) & 127;
    float4 v = ((float4*)agg)[i];
    v.x = fmaxf(v.x * scale[f0 + 0] + shift[f0 + 0], 0.0f);
    v.y = fmaxf(v.y * scale[f0 + 1] + shift[f0 + 1], 0.0f);
    v.z = fmaxf(v.z * scale[f0 + 2] + shift[f0 + 2], 0.0f);
    v.w = fmaxf(v.w * scale[f0 + 3] + shift[f0 + 3], 0.0f);
    ((float4*)agg)[i] = v;
  }
}

// ---- layer 3: t[row] = dot(h2[row], W3) ; one wave per row ----
__global__ __launch_bounds__(256) void k_rowdot(const float* __restrict__ h,
                                                const float* __restrict__ W3,
                                                float* __restrict__ t, int Mrows) {
  int gid = blockIdx.x * 256 + threadIdx.x;
  int r = gid >> 5;
  if (r >= Mrows) return;
  int lane = gid & 31;
  float4 v = ((const float4*)(h + (size_t)r * FEAT))[lane];
  float4 w = ((const float4*)W3)[lane];
  float d = v.x * w.x + v.y * w.y + v.z * w.z + v.w * w.w;
#pragma unroll
  for (int off = 16; off; off >>= 1) d += __shfl_down(d, off, 32);
  if (lane == 0) t[r] = d;
}

__global__ __launch_bounds__(256) void k_init_out(float* __restrict__ out,
                                                  const float* __restrict__ b3, int M) {
  int i = blockIdx.x * 256 + threadIdx.x;
  if (i < M) out[i] = b3[0];
}

__global__ __launch_bounds__(256) void k_scatter_scalar(const float* __restrict__ t,
                                                        const int* __restrict__ ei,
                                                        float* __restrict__ out,
                                                        int E, int total) {
  int gid = blockIdx.x * 256 + threadIdx.x;
  if (gid >= total) return;
  int b = gid / E;
  int e = gid - b * E;
  atomicAdd(&out[(size_t)b * NNODES + ei[E + e]], t[(size_t)b * NNODES + ei[e]]);
}

extern "C" void kernel_launch(void* const* d_in, const int* in_sizes, int n_in,
                              void* d_out, int out_size, void* d_ws, size_t ws_size,
                              hipStream_t stream) {
  const float* x   = (const float*)d_in[0];
  const int*   ei  = (const int*)d_in[1];
  const float* W1  = (const float*)d_in[2];
  const float* b1  = (const float*)d_in[3];
  const float* W2  = (const float*)d_in[4];
  const float* b2  = (const float*)d_in[5];
  const float* W3  = (const float*)d_in[6];
  const float* b3  = (const float*)d_in[7];
  const float* g1  = (const float*)d_in[8];
  const float* be1 = (const float*)d_in[9];
  const float* g2  = (const float*)d_in[10];
  const float* be2 = (const float*)d_in[11];

  const int B = in_sizes[0] / (NNODES * FEAT);
  const int E = in_sizes[1] / 2;
  const int M = B * NNODES;
  const int NF  = M * FEAT;
  const int NF4 = NF / 4;

  // workspace carve (256B aligned)
  char* ws = (char*)d_ws;
  size_t off = 0;
  auto carve = [&](size_t bytes) {
    size_t o = off;
    off = (off + bytes + 255) & ~(size_t)255;
    return o;
  };
  unsigned short* bufA  = (unsigned short*)(ws + carve((size_t)NF * 2)); // bf16 GEMM input
  float*          sup   = (float*)(ws + carve((size_t)NF * 4));
  float*          agg   = (float*)(ws + carve((size_t)NF * 4));
  float*          tvec  = (float*)(ws + carve((size_t)M * 4));
  unsigned short* W1t   = (unsigned short*)(ws + carve(FEAT * FEAT * 2));
  unsigned short* W2t   = (unsigned short*)(ws + carve(FEAT * FEAT * 2));
  float*          gsum  = (float*)(ws + carve(FEAT * 4));
  float*          gsq   = (float*)(ws + carve(FEAT * 4));
  float*          scale = (float*)(ws + carve(FEAT * 4));
  float*          shift = (float*)(ws + carve(FEAT * 4));
  (void)ws_size; (void)n_in; (void)out_size;

  float* out = (float*)d_out;
  const float invM = 1.0f / (float)M;

  const int gW    = (FEAT * FEAT + 255) / 256;
  const int gElem = (NF4 + 255) / 256;
  const int rowTiles = (M + 15) / 16;
  const int gGemm = (rowTiles + 7) / 8;
  const int totalEW = B * E;
  const int gScatV = (totalEW * 32 + 255) / 256;
  const int gStats = 512;
  const int gRow   = (M * 32 + 255) / 256;
  const int gOut   = (M + 255) / 256;
  const int gScatS = (totalEW + 255) / 256;

  // prep
  k_wt_bf16<<<gW, 256, 0, stream>>>(W1, W1t);
  k_wt_bf16<<<gW, 256, 0, stream>>>(W2, W2t);
  k_f32_to_bf16<<<gElem, 256, 0, stream>>>(x, bufA, NF4);

  // ---- layer 1 ----
  k_gemm_wmma<<<gGemm, 256, 0, stream>>>(bufA, W1t, sup, M);
  k_init_bias<<<gElem, 256, 0, stream>>>(agg, b1, NF4);
  k_scatter_vec<<<gScatV, 256, 0, stream>>>(sup, ei, agg, E, totalEW);
  k_zero_stats<<<1, 256, 0, stream>>>(gsum, gsq);
  k_bn_stats<<<gStats, 256, 0, stream>>>(agg, gsum, gsq, NF);
  k_bn_finalize<<<1, 128, 0, stream>>>(gsum, gsq, g1, be1, scale, shift, invM);
  k_bn_relu_bf16<<<gElem, 256, 0, stream>>>(agg, scale, shift, bufA, NF4);

  // ---- layer 2 ----
  k_gemm_wmma<<<gGemm, 256, 0, stream>>>(bufA, W2t, sup, M);
  k_init_bias<<<gElem, 256, 0, stream>>>(agg, b2, NF4);
  k_scatter_vec<<<gScatV, 256, 0, stream>>>(sup, ei, agg, E, totalEW);
  k_zero_stats<<<1, 256, 0, stream>>>(gsum, gsq);
  k_bn_stats<<<gStats, 256, 0, stream>>>(agg, gsum, gsq, NF);
  k_bn_finalize<<<1, 128, 0, stream>>>(gsum, gsq, g2, be2, scale, shift, invM);
  k_bn_relu_f32<<<gElem, 256, 0, stream>>>(agg, scale, shift, NF4);

  // ---- layer 3 (H -> 1) ----
  k_rowdot<<<gRow, 256, 0, stream>>>(agg, W3, tvec, M);
  k_init_out<<<gOut, 256, 0, stream>>>(out, b3, M);
  k_scatter_scalar<<<gScatS, 256, 0, stream>>>(tvec, ei, out, E, totalEW);
}
